// BiLSTMCRF_75058848465621
// MI455X (gfx1250) — matrix operand order repaired and checked by reading, not compiled, not run on hardware
//
#include <hip/hip_runtime.h>

#define BB 1024
#define SS 512
#define LL 64

#define LOG2E 1.4426950408889634f
#define LN2   0.6931471805599453f

typedef __attribute__((ext_vector_type(16))) _Float16 v16h;
typedef __attribute__((ext_vector_type(2)))  _Float16 h2;
typedef __attribute__((ext_vector_type(8)))  float    v8f;
typedef __attribute__((ext_vector_type(8)))  unsigned v8u;

// pack two f32 -> one VGPR of two f16 (RNE); h2 lets the backend write vN.l/vN.h
static __device__ __forceinline__ unsigned pack2h(float a, float b) {
    h2 v;
    v[0] = (_Float16)a;
    v[1] = (_Float16)b;
    return __builtin_bit_cast(unsigned, v);
}
static __device__ __forceinline__ unsigned pack2hh(_Float16 a, _Float16 b) {
    h2 v;
    v[0] = a;
    v[1] = b;
    return __builtin_bit_cast(unsigned, v);
}
// lane <-> lane^16 exchange, no LDS memory touched (ds_swizzle group-of-32, xor=0x10)
static __device__ __forceinline__ unsigned swapx16_u(unsigned x) {
    return (unsigned)__builtin_amdgcn_ds_swizzle((int)x, 0x401F);
}
static __device__ __forceinline__ float swapx16_f(float x) {
    return __int_as_float(__builtin_amdgcn_ds_swizzle(__float_as_int(x), 0x401F));
}
// WGP-scope prefetch (scope field defaults to 0 = WGP -> fills near caches too)
static __device__ __forceinline__ void prefetch_wgp(const void* p) {
    asm volatile("global_prefetch_b8 %0, off" :: "v"(p));
}
static __device__ __forceinline__ v16h as_v16h(const unsigned x[8]) {
    v8u u = { x[0], x[1], x[2], x[3], x[4], x[5], x[6], x[7] };
    return __builtin_bit_cast(v16h, u);
}
static __device__ __forceinline__ v8f wmma_acc(const unsigned a[8], v16h B, v8f C) {
    return __builtin_amdgcn_wmma_f32_16x16x32_f16(
        /*neg_a=*/false, as_v16h(a), /*neg_b=*/false, B,
        /*c_mod=*/(short)0, C, /*reuse_a=*/false, /*reuse_b=*/false);
}
static __device__ __forceinline__ void load8(const float* p, float d[8]) {
    const float4* p4 = reinterpret_cast<const float4*>(p);
    float4 a = p4[0], b = p4[1];
    d[0]=a.x; d[1]=a.y; d[2]=a.z; d[3]=a.w;
    d[4]=b.x; d[5]=b.y; d[6]=b.z; d[7]=b.w;
}

// ---------------------------------------------------------------------------
// Forward (alpha) scan. One wave32 per 16 batch rows. Grid = B/16 = 64 blocks.
// Score tile lives in registers in WMMA C/D layout: D[j, b] with
//   b = lane&15, j = 16*mt + r + 8*(lane>>4),  mt=0..3 (j-tiles), r=0..7 (VGPR).
// A operand = exp(transitions) (constant, f16 hi+lo split, resident in VGPRs).
// B operand = exp(score - rowmax) rebuilt each step via pack + SWAPX16 swizzle.
// Fully branchless step body: no EXEC manipulation, raw v_exp/v_log transcendentals.
// ---------------------------------------------------------------------------
__global__ __launch_bounds__(32)
void crf_forward_kernel(const float* __restrict__ emis,
                        const unsigned char* __restrict__ mask,
                        const float* __restrict__ trans,
                        const float* __restrict__ startT,
                        const float* __restrict__ endT,
                        float* __restrict__ fwd_out) {
    const int lane = threadIdx.x;         // 0..31
    const int c    = lane & 15;           // batch column within tile
    const int h    = lane >> 4;           // lane half
    const bool hiH = (h != 0);
    const int b    = blockIdx.x * 16 + c;
    const float* erow = emis + (size_t)b * SS * LL;
    const unsigned char* mrow = mask + (size_t)b * SS;

    // ---- constant A operands: Ehat[j][i] = exp(transitions[j][i]), hi/lo f16 split
    // A packing used here (16x32, contiguous k, same as B): lane%16 = M(=j); VGPR q: K = 16h + 2q + s
    unsigned ehiP[4][2][8], eloP[4][2][8];
#pragma unroll
    for (int mt = 0; mt < 4; ++mt) {
        const int j = 16 * mt + c;
        const float* tr = trans + j * LL;
#pragma unroll
        for (int kh = 0; kh < 2; ++kh) {
#pragma unroll
            for (int q = 0; q < 8; ++q) {
                const int i0 = kh * 32 + 16 * h + 2 * q;
                float e0 = __builtin_amdgcn_exp2f(tr[i0]     * LOG2E);
                float e1 = __builtin_amdgcn_exp2f(tr[i0 + 1] * LOG2E);
                _Float16 h0 = (_Float16)e0, h1 = (_Float16)e1;
                float l0 = e0 - (float)h0, l1 = e1 - (float)h1;
                ehiP[mt][kh][q] = pack2hh(h0, h1);
                eloP[mt][kh][q] = pack2h(l0, l1);
            }
        }
    }

    // ---- score0 = start + emissions[:,0,:]
    float sc[4][8];
#pragma unroll
    for (int mt = 0; mt < 4; ++mt) {
        const int j0 = 16 * mt + 8 * h;
        float ev[8];
        load8(erow + j0, ev);
#pragma unroll
        for (int r = 0; r < 8; ++r) sc[mt][r] = startT[j0 + r] + ev[r];
    }

    // ---- scan over time
    for (int t = 1; t < SS; ++t) {
        // unconditional clamped look-ahead prefetch into WGP-near caches
        {
            int tp = t + 6; tp = (tp < SS) ? tp : (SS - 1);
            prefetch_wgp(erow + (size_t)tp * LL + 32 * h);
        }

        // emissions tile for this step (8 contiguous floats per (mt,half))
        float em[4][8];
#pragma unroll
        for (int mt = 0; mt < 4; ++mt)
            load8(erow + (size_t)t * LL + 16 * mt + 8 * h, em[mt]);
        const float mkf = mrow[t] ? 1.0f : 0.0f;   // branchless keep/update factor
        const float omk = 1.0f - mkf;

        // per-row max over all 64 j: 5-level tree in-lane + half swap
        float tm[16];
#pragma unroll
        for (int k = 0; k < 16; ++k)
            tm[k] = fmaxf(sc[k >> 2][(k & 3) * 2], sc[k >> 2][(k & 3) * 2 + 1]);
#pragma unroll
        for (int k = 0; k < 8; ++k) tm[k] = fmaxf(tm[k], tm[k + 8]);
#pragma unroll
        for (int k = 0; k < 4; ++k) tm[k] = fmaxf(tm[k], tm[k + 4]);
        tm[0] = fmaxf(fmaxf(tm[0], tm[2]), fmaxf(tm[1], tm[3]));
        const float mx   = fmaxf(tm[0], swapx16_f(tm[0]));
        const float nmxl = -mx * LOG2E;            // folded into per-element FMA

        // p = exp2(sc*log2e - mx*log2e), packed to f16 pairs; other half via swizzle
        unsigned pk[4][4], sw[4][4];
#pragma unroll
        for (int mt = 0; mt < 4; ++mt)
#pragma unroll
            for (int a = 0; a < 4; ++a) {
                float p0 = __builtin_amdgcn_exp2f(fmaf(sc[mt][2 * a],     LOG2E, nmxl));
                float p1 = __builtin_amdgcn_exp2f(fmaf(sc[mt][2 * a + 1], LOG2E, nmxl));
                pk[mt][a] = pack2h(p0, p1);
                sw[mt][a] = swapx16_u(pk[mt][a]);
            }

        // assemble B operands (32x16): lane%16 = N(=b); VGPR q: K = 16h + 2q, 2q+1
        unsigned b0[8], b1[8];
#pragma unroll
        for (int q = 0; q < 4; ++q) {
            b0[q]     = hiH ? sw[1][q] : pk[0][q];
            b0[q + 4] = hiH ? pk[1][q] : sw[0][q];
            b1[q]     = hiH ? sw[3][q] : pk[2][q];
            b1[q + 4] = hiH ? pk[3][q] : sw[2][q];
        }
        const v16h B0 = as_v16h(b0);
        const v16h B1 = as_v16h(b1);

        // D[j,b] = Ehat x p ; hi/lo split for ~fp32-accurate E
#pragma unroll
        for (int mt = 0; mt < 4; ++mt) {
            v8f acc = {};
            acc = wmma_acc(eloP[mt][1], B1, acc);
            acc = wmma_acc(ehiP[mt][1], B1, acc);
            acc = wmma_acc(eloP[mt][0], B0, acc);
            acc = wmma_acc(ehiP[mt][0], B0, acc);
#pragma unroll
            for (int r = 0; r < 8; ++r) {
                // acc >= ~0.9 always (row max included) -> raw v_log_f32 is safe
                float lg = __builtin_amdgcn_logf(acc[r]);          // log2(acc)
                float nv = fmaf(lg, LN2, em[mt][r] + mx);
                // exact blend: mkf==1 -> nv, mkf==0 -> old score; no branches
                sc[mt][r] = fmaf(mkf, nv, omk * sc[mt][r]);
            }
        }
    }

    // ---- forward_score[b] = logsumexp_j(score + end)
    float v[4][8];
#pragma unroll
    for (int mt = 0; mt < 4; ++mt)
#pragma unroll
        for (int r = 0; r < 8; ++r)
            v[mt][r] = sc[mt][r] + endT[16 * mt + 8 * h + r];

    float tm[16];
#pragma unroll
    for (int k = 0; k < 16; ++k)
        tm[k] = fmaxf(v[k >> 2][(k & 3) * 2], v[k >> 2][(k & 3) * 2 + 1]);
#pragma unroll
    for (int k = 0; k < 8; ++k) tm[k] = fmaxf(tm[k], tm[k + 8]);
#pragma unroll
    for (int k = 0; k < 4; ++k) tm[k] = fmaxf(tm[k], tm[k + 4]);
    tm[0] = fmaxf(fmaxf(tm[0], tm[2]), fmaxf(tm[1], tm[3]));
    const float m2   = fmaxf(tm[0], swapx16_f(tm[0]));
    const float nm2l = -m2 * LOG2E;

    float s = 0.f;
#pragma unroll
    for (int mt = 0; mt < 4; ++mt)
#pragma unroll
        for (int r = 0; r < 8; ++r)
            s += __builtin_amdgcn_exp2f(fmaf(v[mt][r], LOG2E, nm2l));
    s += swapx16_f(s);
    float fs = fmaf(__builtin_amdgcn_logf(s), LN2, m2);
    if (lane < 16) fwd_out[blockIdx.x * 16 + c] = fs;
}

// ---------------------------------------------------------------------------
// Gold path score: one wave per batch row (gathered loads + wave reduction).
// ---------------------------------------------------------------------------
__global__ __launch_bounds__(256)
void crf_gold_kernel(const float* __restrict__ emis,
                     const int* __restrict__ tags,
                     const unsigned char* __restrict__ mask,
                     const float* __restrict__ trans,
                     const float* __restrict__ startT,
                     const float* __restrict__ endT,
                     float* __restrict__ gold_out) {
    const int wid  = threadIdx.x >> 5;
    const int lane = threadIdx.x & 31;
    const int b    = blockIdx.x * 8 + wid;
    const int* trow = tags + (size_t)b * SS;
    const unsigned char* mrow = mask + (size_t)b * SS;
    const float* erow = emis + (size_t)b * SS * LL;

    float acc = 0.f;
    int   len = 0;
    for (int t = lane; t < SS; t += 32) {
        int tg = trow[t];
        int mk = mrow[t] ? 1 : 0;
        float em = erow[(size_t)t * LL + tg];
        if (t == 0) {
            acc += em;                       // emit at t=0, unconditional
        } else if (mk) {
            acc += trans[tg * LL + trow[t - 1]] + em;
        }
        len += mk;
    }
#pragma unroll
    for (int k = 16; k >= 1; k >>= 1) {
        acc += __shfl_xor(acc, k, 32);
        len += __shfl_xor(len, k, 32);
    }
    if (lane == 0) {
        int t0 = trow[0];
        int lastT = trow[len - 1];
        gold_out[b] = startT[t0] + acc + endT[lastT];
    }
}

// ---------------------------------------------------------------------------
// Final mean(forward - gold) -> single f32
// ---------------------------------------------------------------------------
__global__ __launch_bounds__(256)
void crf_reduce_kernel(const float* __restrict__ fwd,
                       const float* __restrict__ gold,
                       float* __restrict__ out) {
    __shared__ float sm[256];
    float s = 0.f;
    for (int i = threadIdx.x; i < BB; i += 256) s += fwd[i] - gold[i];
    sm[threadIdx.x] = s;
    __syncthreads();
    for (int k = 128; k > 0; k >>= 1) {
        if (threadIdx.x < k) sm[threadIdx.x] += sm[threadIdx.x + k];
        __syncthreads();
    }
    if (threadIdx.x == 0) out[0] = sm[0] / (float)BB;
}

extern "C" void kernel_launch(void* const* d_in, const int* in_sizes, int n_in,
                              void* d_out, int out_size, void* d_ws, size_t ws_size,
                              hipStream_t stream) {
    const float*         emis   = (const float*)d_in[0];
    const int*           tags   = (const int*)d_in[1];
    const unsigned char* mask   = (const unsigned char*)d_in[2];  // jnp bool = 1 byte
    const float*         trans  = (const float*)d_in[3];
    const float*         startT = (const float*)d_in[4];
    const float*         endT   = (const float*)d_in[5];

    float* fwd  = (float*)d_ws;
    float* gold = fwd + BB;

    crf_forward_kernel<<<dim3(BB / 16), dim3(32), 0, stream>>>(
        emis, mask, trans, startT, endT, fwd);
    crf_gold_kernel<<<dim3(BB / 8), dim3(256), 0, stream>>>(
        emis, tags, mask, trans, startT, endT, gold);
    crf_reduce_kernel<<<1, 256, 0, stream>>>(fwd, gold, (float*)d_out);
}
